// MainGNNModel_32822140076341
// MI455X (gfx1250) — matrix operand order repaired and checked by reading, not compiled, not run on hardware
//
#include <hip/hip_runtime.h>

#define N_PFAS 20000
#define N_GW   100000
#define N_SW   20000
#define D_IN   64
#define D_OUT  128
#define E_PG   1600000
#define E_GP   1600000
#define E_SP   400000
#define E_PS   400000

typedef __attribute__((ext_vector_type(16))) _Float16 v16h;
typedef __attribute__((ext_vector_type(8)))  _Float16 v8h;
typedef __attribute__((ext_vector_type(8)))  float    v8f;

// ---------------------------------------------------------------------------
// Zero a float region (grid-stride).
// ---------------------------------------------------------------------------
__global__ void zero_f32(float* __restrict__ p, long long n) {
  long long i = (long long)blockIdx.x * blockDim.x + threadIdx.x;
  long long stride = (long long)gridDim.x * blockDim.x;
  for (; i < n; i += stride) p[i] = 0.0f;
}

// ---------------------------------------------------------------------------
// Edge scatter: 16 threads per edge, each handles 4 contiguous features.
// agg[dst] += x_src[src]; cnt[dst] += 1 (chunk 0 only).
// ---------------------------------------------------------------------------
__global__ void scatter_edges(const float* __restrict__ xsrc,
                              const int* __restrict__ src,
                              const int* __restrict__ dst,
                              float* __restrict__ agg,
                              float* __restrict__ cnt, int E) {
  long long t = (long long)blockIdx.x * blockDim.x + threadIdx.x;
  int e = (int)(t >> 4);
  int c = (int)(t & 15);
  if (e >= E) return;
  int s = src[e];
  int d = dst[e];
  const float4 v = *(const float4*)(xsrc + (size_t)s * D_IN + c * 4);
  float* ap = agg + (size_t)d * D_IN + c * 4;
  atomicAdd(ap + 0, v.x);
  atomicAdd(ap + 1, v.y);
  atomicAdd(ap + 2, v.z);
  atomicAdd(ap + 3, v.w);
  if (c == 0) atomicAdd(cnt + d, 1.0f);
}

// ---------------------------------------------------------------------------
// Pack [Wl ; Wr] (each 64x128 f32 row-major) into one 128x128 f16 K-major mat.
// ---------------------------------------------------------------------------
__global__ void pack_w2(const float* __restrict__ Wl, const float* __restrict__ Wr,
                        _Float16* __restrict__ out) {
  int idx = blockIdx.x * blockDim.x + threadIdx.x;
  if (idx >= 128 * 128) return;
  float v = (idx < 64 * 128) ? Wl[idx] : Wr[idx - 64 * 128];
  out[idx] = (_Float16)v;
}

// ---------------------------------------------------------------------------
// Pack pfas weights: [Wl_gp ; Wl_sp ; Wr_gp+Wr_sp] -> 192x128 f16,
// and bias = b_gp + b_sp.
// ---------------------------------------------------------------------------
__global__ void pack_w_pfas(const float* __restrict__ Wl_gp, const float* __restrict__ Wl_sp,
                            const float* __restrict__ Wr_gp, const float* __restrict__ Wr_sp,
                            const float* __restrict__ b_gp, const float* __restrict__ b_sp,
                            _Float16* __restrict__ out, float* __restrict__ bias_out) {
  int idx = blockIdx.x * blockDim.x + threadIdx.x;
  if (idx < 192 * 128) {
    float v;
    if (idx < 64 * 128)       v = Wl_gp[idx];
    else if (idx < 128 * 128) v = Wl_sp[idx - 64 * 128];
    else                      v = Wr_gp[idx - 128 * 128] + Wr_sp[idx - 128 * 128];
    out[idx] = (_Float16)v;
  }
  if (idx < 128) bias_out[idx] = b_gp[idx] + b_sp[idx];
}

// ---------------------------------------------------------------------------
// Build f16 A matrix: [mean1 | (mean2) | x_root], row-major n x K.
// K == 128: cols 0-63 mean1, 64-127 x.  K == 192: mean1, mean2, x.
// ---------------------------------------------------------------------------
__global__ void build_A(const float* __restrict__ agg1, const float* __restrict__ cnt1,
                        const float* __restrict__ agg2, const float* __restrict__ cnt2,
                        const float* __restrict__ x, _Float16* __restrict__ A,
                        int n, int K) {
  long long idx = (long long)blockIdx.x * blockDim.x + threadIdx.x;
  long long total = (long long)n * K;
  if (idx >= total) return;
  int row = (int)(idx / K);
  int c   = (int)(idx % K);
  float v;
  if (c < 64) {
    v = agg1[(size_t)row * 64 + c] / fmaxf(cnt1[row], 1.0f);
  } else if (K == 192 && c < 128) {
    v = agg2[(size_t)row * 64 + (c - 64)] / fmaxf(cnt2[row], 1.0f);
  } else {
    v = x[(size_t)row * 64 + (c - (K - 64))];
  }
  A[idx] = (_Float16)v;
}

// ---------------------------------------------------------------------------
// Shared WMMA core: block = 256 threads = 8 waves, one 16-row M tile per
// block staged in LDS; wave w owns N tile w (16 cols).  Per-lane WMMA
// operand layouts per ISA 7.12.2 (wave32):
//   A 16x32: lane l -> row l&15; VGPR0-3 = K kb..kb+7, VGPR4-7 = kb+16..23
//   B 32x16: lane l -> col n0+(l&15); VGPR0-3 = K kb..kb+7, VGPR4-7 = kb+16..23
//   C/D:     VGPR j -> [M = 8*(lane>>4)+j][N = lane&15]
// ---------------------------------------------------------------------------
template <int K>
__device__ __forceinline__ v8f wmma_tile(const _Float16* ldsA,
                                         const _Float16* __restrict__ W,
                                         int lane, int n0) {
  const int kb = (lane >> 4) * 8;
  const _Float16* ap = ldsA + (size_t)(lane & 15) * K;
  v8f acc = {};
#pragma unroll
  for (int ks = 0; ks < K; ks += 32) {
    v8h alo = *(const v8h*)(ap + ks + kb);        // ds_load_b128
    v8h ahi = *(const v8h*)(ap + ks + kb + 16);   // ds_load_b128
    v16h a;
#pragma unroll
    for (int i = 0; i < 8; ++i) { a[i] = alo[i]; a[i + 8] = ahi[i]; }
    v16h b;
#pragma unroll
    for (int i = 0; i < 8; ++i) {
      b[i]     = W[(size_t)(ks + kb + i) * D_OUT + n0 + (lane & 15)];
      b[i + 8] = W[(size_t)(ks + kb + 16 + i) * D_OUT + n0 + (lane & 15)];
    }
    acc = __builtin_amdgcn_wmma_f32_16x16x32_f16(
        false, a, false, b, (short)0, acc, false, false);
  }
  return acc;
}

template <int K>
__device__ __forceinline__ void stage_A(const _Float16* __restrict__ A,
                                        _Float16* ldsA, int m0) {
  const _Float16* gsrc = A + (size_t)m0 * K;
#pragma unroll
  for (int t = threadIdx.x; t < (16 * K) / 8; t += 256)
    *(v8h*)(ldsA + t * 8) = *(const v8h*)(gsrc + t * 8);   // coalesced 16B
  __syncthreads();
}

// ---------------------------------------------------------------------------
// pfas GEMM: out_h[n x 128] = relu(A @ W + bias)   (h_pfas is a model output)
// ---------------------------------------------------------------------------
template <int K>
__global__ __launch_bounds__(256)
void gemm_bias_relu(const _Float16* __restrict__ A, const _Float16* __restrict__ W,
                    const float* __restrict__ bias, float* __restrict__ out) {
  __shared__ _Float16 ldsA[16 * K];
  const int lane = threadIdx.x & 31;
  const int wave = threadIdx.x >> 5;
  const int m0   = blockIdx.x * 16;
  const int n0   = wave * 16;

  stage_A<K>(A, ldsA, m0);
  v8f acc = wmma_tile<K>(ldsA, W, lane, n0);

  const int colN = n0 + (lane & 15);
  const int mhi  = (lane >> 4) * 8;
  const float bv = bias[colN];
#pragma unroll
  for (int j = 0; j < 8; ++j) {
    float v = acc[j] + bv;
    v = v > 0.0f ? v : 0.0f;
    out[(size_t)(m0 + mhi + j) * D_OUT + colN] = v;
  }
}

// ---------------------------------------------------------------------------
// gw/sw GEMM with fused 128->1 PReLU head:
//   out[i] = prelu( relu(A@W + bias)[i,:] . W_lin + b_lin )
// h never touches HBM: contributions h*W_lin[col] staged in LDS (stride 132
// to break bank conflicts) and reduced in two stages.
// ---------------------------------------------------------------------------
#define HSTRIDE 132
template <int K>
__global__ __launch_bounds__(256)
void gemm_relu_head(const _Float16* __restrict__ A, const _Float16* __restrict__ W,
                    const float* __restrict__ bias, const float* __restrict__ Wlin,
                    const float* __restrict__ blin, const float* __restrict__ alpha,
                    float* __restrict__ out) {
  __shared__ _Float16 ldsA[16 * K];
  __shared__ float hbuf[16 * HSTRIDE];
  __shared__ float red[16 * 8];
  const int lane = threadIdx.x & 31;
  const int wave = threadIdx.x >> 5;
  const int m0   = blockIdx.x * 16;
  const int n0   = wave * 16;

  stage_A<K>(A, ldsA, m0);
  v8f acc = wmma_tile<K>(ldsA, W, lane, n0);

  const int colN = n0 + (lane & 15);
  const int mhi  = (lane >> 4) * 8;
  const float bv = bias[colN];
  const float wl = Wlin[colN];
#pragma unroll
  for (int j = 0; j < 8; ++j) {
    float v = acc[j] + bv;
    v = v > 0.0f ? v : 0.0f;           // relu(h)
    hbuf[(mhi + j) * HSTRIDE + colN] = v * wl;
  }
  __syncthreads();

  if (threadIdx.x < 128) {             // stage 1: 8 partials per row
    int row = threadIdx.x >> 3, ch = threadIdx.x & 7;
    const float* hp = hbuf + row * HSTRIDE + ch * 16;
    float s = 0.0f;
#pragma unroll
    for (int j = 0; j < 16; ++j) s += hp[j];
    red[row * 8 + ch] = s;
  }
  __syncthreads();

  if (threadIdx.x < 16) {              // stage 2: finish dot + PReLU
    float s = blin[0];
#pragma unroll
    for (int j = 0; j < 8; ++j) s += red[threadIdx.x * 8 + j];
    out[m0 + threadIdx.x] = s > 0.0f ? s : alpha[0] * s;
  }
}

// ---------------------------------------------------------------------------
extern "C" void kernel_launch(void* const* d_in, const int* in_sizes, int n_in,
                              void* d_out, int out_size, void* d_ws, size_t ws_size,
                              hipStream_t stream) {
  (void)in_sizes; (void)n_in; (void)out_size; (void)ws_size;

  const float* x_pfas = (const float*)d_in[0];
  const float* x_gw   = (const float*)d_in[1];
  const float* x_sw   = (const float*)d_in[2];
  const int* ei_pg_src = (const int*)d_in[3];
  const int* ei_pg_dst = (const int*)d_in[4];
  const int* ei_gp_src = (const int*)d_in[5];
  const int* ei_gp_dst = (const int*)d_in[6];
  const int* ei_sp_src = (const int*)d_in[7];
  const int* ei_sp_dst = (const int*)d_in[8];
  const int* ei_ps_src = (const int*)d_in[9];
  const int* ei_ps_dst = (const int*)d_in[10];
  const float* Wl_pg = (const float*)d_in[11];
  const float* Wr_pg = (const float*)d_in[12];
  const float* b_pg  = (const float*)d_in[13];
  const float* Wl_gp = (const float*)d_in[14];
  const float* Wr_gp = (const float*)d_in[15];
  const float* b_gp  = (const float*)d_in[16];
  const float* Wl_sp = (const float*)d_in[17];
  const float* Wr_sp = (const float*)d_in[18];
  const float* b_sp  = (const float*)d_in[19];
  const float* Wl_ps = (const float*)d_in[20];
  const float* Wr_ps = (const float*)d_in[21];
  const float* b_ps  = (const float*)d_in[22];
  const float* W_lin = (const float*)d_in[23];
  const float* b_lin = (const float*)d_in[24];
  const float* alpha = (const float*)d_in[25];

  float* out       = (float*)d_out;
  float* out_hpfas = out;                              // [N_PFAS, 128]
  float* out_gw    = out + (size_t)N_PFAS * D_OUT;     // [N_GW]
  float* out_sw    = out_gw + N_GW;                    // [N_SW]

  // ---- workspace carve-out (256B aligned) ----
  char* ws = (char*)d_ws;
  size_t off = 0;
  auto alloc = [&](size_t bytes) -> void* {
    off = (off + 255) & ~(size_t)255;
    void* p = ws + off;
    off += bytes;
    return p;
  };

  float* agg_pg = (float*)alloc((size_t)N_GW * 64 * 4);
  float* cnt_pg = (float*)alloc((size_t)N_GW * 4);
  float* agg_gp = (float*)alloc((size_t)N_PFAS * 64 * 4);
  float* cnt_gp = (float*)alloc((size_t)N_PFAS * 4);
  float* agg_sp = (float*)alloc((size_t)N_PFAS * 64 * 4);
  float* cnt_sp = (float*)alloc((size_t)N_PFAS * 4);
  float* agg_ps = (float*)alloc((size_t)N_SW * 64 * 4);
  float* cnt_ps = (float*)alloc((size_t)N_SW * 4);
  size_t zero_end = off;                 // zero everything from agg_pg to here
  _Float16* A_gw   = (_Float16*)alloc((size_t)N_GW * 128 * 2);
  _Float16* A_pfas = (_Float16*)alloc((size_t)N_PFAS * 192 * 2);
  _Float16* A_sw   = (_Float16*)alloc((size_t)N_SW * 128 * 2);
  _Float16* W_gw   = (_Float16*)alloc(128 * 128 * 2);
  _Float16* W_pfas = (_Float16*)alloc(192 * 128 * 2);
  _Float16* W_sw   = (_Float16*)alloc(128 * 128 * 2);
  float* bias_pfas = (float*)alloc(128 * 4);

  // ---- 1. zero aggregation buffers ----
  long long zero_floats = (long long)(((char*)ws + zero_end) - (char*)agg_pg) / 4;
  zero_f32<<<2048, 256, 0, stream>>>(agg_pg, zero_floats);

  // ---- 2. edge scatter (atomic segment-sum + counts) ----
  auto sblocks = [](long long E) { return (unsigned)((E * 16 + 255) / 256); };
  scatter_edges<<<sblocks(E_PG), 256, 0, stream>>>(x_pfas, ei_pg_src, ei_pg_dst, agg_pg, cnt_pg, E_PG);
  scatter_edges<<<sblocks(E_GP), 256, 0, stream>>>(x_gw,   ei_gp_src, ei_gp_dst, agg_gp, cnt_gp, E_GP);
  scatter_edges<<<sblocks(E_SP), 256, 0, stream>>>(x_sw,   ei_sp_src, ei_sp_dst, agg_sp, cnt_sp, E_SP);
  scatter_edges<<<sblocks(E_PS), 256, 0, stream>>>(x_pfas, ei_ps_src, ei_ps_dst, agg_ps, cnt_ps, E_PS);

  // ---- 3. pack weights to f16 (K-major), fuse Wr_gp+Wr_sp and b_gp+b_sp ----
  pack_w2<<<(128 * 128 + 255) / 256, 256, 0, stream>>>(Wl_pg, Wr_pg, W_gw);
  pack_w2<<<(128 * 128 + 255) / 256, 256, 0, stream>>>(Wl_ps, Wr_ps, W_sw);
  pack_w_pfas<<<(192 * 128 + 255) / 256, 256, 0, stream>>>(
      Wl_gp, Wl_sp, Wr_gp, Wr_sp, b_gp, b_sp, W_pfas, bias_pfas);

  // ---- 4. build f16 A matrices [mean | (mean2) | x_root] ----
  {
    long long t = (long long)N_GW * 128;
    build_A<<<(unsigned)((t + 255) / 256), 256, 0, stream>>>(
        agg_pg, cnt_pg, nullptr, nullptr, x_gw, A_gw, N_GW, 128);
  }
  {
    long long t = (long long)N_PFAS * 192;
    build_A<<<(unsigned)((t + 255) / 256), 256, 0, stream>>>(
        agg_gp, cnt_gp, agg_sp, cnt_sp, x_pfas, A_pfas, N_PFAS, 192);
  }
  {
    long long t = (long long)N_SW * 128;
    build_A<<<(unsigned)((t + 255) / 256), 256, 0, stream>>>(
        agg_ps, cnt_ps, nullptr, nullptr, x_sw, A_sw, N_SW, 128);
  }

  // ---- 5. WMMA GEMMs ----
  // pfas: h is an output -> plain GEMM+bias+ReLU to d_out
  gemm_bias_relu<192><<<N_PFAS / 16, 256, 0, stream>>>(A_pfas, W_pfas, bias_pfas, out_hpfas);
  // gw/sw: fuse the 128->1 PReLU head; h never hits HBM
  gemm_relu_head<128><<<N_GW / 16, 256, 0, stream>>>(A_gw, W_gw, b_pg, W_lin, b_lin, alpha, out_gw);
  gemm_relu_head<128><<<N_SW / 16, 256, 0, stream>>>(A_sw, W_sw, b_ps, W_lin, b_lin, alpha, out_sw);
}
